// KnotGCN_2705829397354
// MI455X (gfx1250) — matrix operand order repaired and checked by reading, not compiled, not run on hardware
//
#include <hip/hip_runtime.h>
#include <hip/hip_bf16.h>

#define NNODES  100000
#define NEDGES  640000
#define HDIM    128
#define NLAYERS 4
#define NCLASS  2
#define NGRAPH  128

typedef __attribute__((ext_vector_type(16))) __bf16 v16bf;
typedef __attribute__((ext_vector_type(8)))  float  v8f;

__device__ __forceinline__ void atomAddF(float* p, float v) {
  unsafeAtomicAdd(p, v);                    // native global_atomic_add_f32
}

// ---------------- small elementwise kernels ----------------

__global__ void k_fill(float* a, float v, int n) {
  int i = blockIdx.x * blockDim.x + threadIdx.x;
  if (i < n) a[i] = v;
}

__global__ void k_count_deg(const int* __restrict__ dst, float* deg, int e) {
  int i = blockIdx.x * blockDim.x + threadIdx.x;
  if (i < e) atomAddF(&deg[dst[i]], 1.0f);
}

__global__ void k_dinv(float* deg, int n) {
  int i = blockIdx.x * blockDim.x + threadIdx.x;
  if (i < n) deg[i] = __frsqrt_rn(deg[i]);   // in place: deg -> dinv
}

__global__ void k_enorm(const int* __restrict__ src, const int* __restrict__ dst,
                        const float* __restrict__ dinv, float* enorm, int e) {
  int i = blockIdx.x * blockDim.x + threadIdx.x;
  if (i < e) enorm[i] = dinv[src[i]] * dinv[dst[i]];
}

__global__ void k_init_h(const int* __restrict__ x, const float* __restrict__ embed,
                         float* h, int nh) {
  int i = blockIdx.x * blockDim.x + threadIdx.x;
  if (i >= nh) return;
  int n = i >> 7, j = i & 127;
  h[i] = embed[x[n] * HDIM + j];
}

// Pre-pack weights into exact B-fragment layout:
// Wfrag[((l*4+kb)*8+nt)*32*16 + lane*16 + i] = bf16(Ws[l][k][n]),
//   n = nt*16 + lane%16,  k = kb*32 + (lane/16)*8 + (i%8) + (i/8)*16
__global__ void k_prep_w(const float* __restrict__ Ws, __bf16* Wfrag) {
  int idx = blockIdx.x * blockDim.x + threadIdx.x;     // L*4*8*32*16 = 65536
  if (idx >= NLAYERS * 4 * 8 * 32 * 16) return;
  int i    = idx & 15;
  int lane = (idx >> 4) & 31;
  int nt   = (idx >> 9) & 7;
  int kb   = (idx >> 12) & 3;
  int l    = idx >> 14;
  int n = nt * 16 + (lane & 15);
  int k = kb * 32 + (lane >> 4) * 8 + (i & 7) + (i >> 3) * 16;
  Wfrag[idx] = (__bf16)Ws[(l * HDIM + k) * HDIM + n];
}

// ---------------- WMMA GEMM: t = (relu?)h @ W   (N x 128) @ (128 x 128) ----------------
// 256 threads = 8 waves; each wave produces a 16x128 output tile.
// RELU is a compile-time template parameter -> single fused max+cvt, no cndmask.

template <bool RELU>
__global__ __launch_bounds__(256) void k_gemm(const float* __restrict__ h,
                                              const __bf16* __restrict__ Wfrag,
                                              float* __restrict__ t, int n) {
  const int lane = threadIdx.x & 31;
  const int wave = threadIdx.x >> 5;
  const int m0   = blockIdx.x * 128 + wave * 16;

  const int rowA  = min(m0 + (lane & 15), n - 1);      // clamp reads; EXEC stays all-ones
  const int kbase = (lane >> 4) * 8;                   // A-layout K interleave
  const float* hrow = h + rowA * HDIM;

  v8f acc[8];
#pragma unroll
  for (int nt = 0; nt < 8; ++nt) acc[nt] = (v8f){0.f,0.f,0.f,0.f,0.f,0.f,0.f,0.f};

#pragma unroll
  for (int kb = 0; kb < 4; ++kb) {
    const int k0 = kb * 32 + kbase;
    v16bf a;
#pragma unroll
    for (int i = 0; i < 8; ++i) {
      float x0 = hrow[k0 + i];                          // K = kbase + i
      float x1 = hrow[k0 + 16 + i];                     // K = kbase + 16 + i
      if (RELU) { x0 = fmaxf(x0, 0.f); x1 = fmaxf(x1, 0.f); }
      a[i]     = (__bf16)x0;                            // native v_cvt_pk_bf16_f32 path
      a[i + 8] = (__bf16)x1;
    }
    const __bf16* wk = Wfrag + (kb * 8) * 32 * 16 + lane * 16;
#pragma unroll
    for (int nt = 0; nt < 8; ++nt) {
      v16bf b = *(const v16bf*)(wk + nt * 32 * 16);
      acc[nt] = __builtin_amdgcn_wmma_f32_16x16x32_bf16(
          false, a, false, b, (short)0, acc[nt], false, false);
    }
  }

  const int rbase = m0 + (lane >> 4) * 8;               // D layout: M = r + lanehalf*8
  const int col   = lane & 15;
#pragma unroll
  for (int nt = 0; nt < 8; ++nt) {
#pragma unroll
    for (int r = 0; r < 8; ++r) {
      int rr = rbase + r;
      if (rr < n) t[rr * HDIM + nt * 16 + col] = acc[nt][r];
    }
  }
}

// agg_init: h_out = t * snorm + b    (snorm = dinv^2)
__global__ void k_init_agg(const float* __restrict__ t, const float* __restrict__ dinv,
                           const float* __restrict__ b, float* hout, int nh) {
  int i = blockIdx.x * blockDim.x + threadIdx.x;
  if (i >= nh) return;
  int n = i >> 7, j = i & 127;
  float dv = dinv[n];
  hout[i] = t[i] * dv * dv + b[j];
}

// edge scatter: hout[dst] += t[src] * enorm   (thread = one edge x 4-float chunk)
__global__ void k_edge_scatter(const int* __restrict__ src, const int* __restrict__ dst,
                               const float* __restrict__ enorm, const float* __restrict__ t,
                               float* hout, int e) {
  int i = blockIdx.x * blockDim.x + threadIdx.x;       // e * 32 threads
  int ed = i >> 5;
  if (ed >= e) return;
  int q = (i & 31) * 4;
  int s = src[ed], d = dst[ed];
  float w = enorm[ed];
  const float4 v = *(const float4*)(t + s * HDIM + q);
  float* o = hout + d * HDIM + q;
  atomAddF(o + 0, v.x * w);
  atomAddF(o + 1, v.y * w);
  atomAddF(o + 2, v.z * w);
  atomAddF(o + 3, v.w * w);
}

// pooling: pooled[g] += h[n]; counts[g] += 1
__global__ void k_pool(const float* __restrict__ h, const int* __restrict__ batch,
                       float* pooled, float* counts, int nh) {
  int i = blockIdx.x * blockDim.x + threadIdx.x;
  if (i >= nh) return;
  int n = i >> 7, j = i & 127;
  int g = batch[n];
  atomAddF(&pooled[g * HDIM + j], h[i]);
  if (j == 0) atomAddF(&counts[g], 1.0f);
}

// out[g][c] = (pooled[g]/max(cnt,1)) . Wout[:,c] + bout[c]   -- 256 threads total
__global__ void k_out(const float* __restrict__ pooled, const float* __restrict__ counts,
                      const float* __restrict__ Wout, const float* __restrict__ bout,
                      float* out) {
  int idx = threadIdx.x;                                // G*C = 256
  int g = idx >> 1, c = idx & 1;
  float cnt = fmaxf(counts[g], 1.0f);
  float s = 0.f;
#pragma unroll 8
  for (int j = 0; j < HDIM; ++j) s += pooled[g * HDIM + j] * Wout[j * NCLASS + c];
  out[g * NCLASS + c] = s / cnt + bout[c];
}

// ---------------- host orchestration ----------------

extern "C" void kernel_launch(void* const* d_in, const int* in_sizes, int n_in,
                              void* d_out, int out_size, void* d_ws, size_t ws_size,
                              hipStream_t stream) {
  const int*   x     = (const int*)d_in[0];
  const int*   ei    = (const int*)d_in[1];
  const int*   src   = ei;
  const int*   dst   = ei + NEDGES;
  const int*   batch = (const int*)d_in[2];
  const float* embed = (const float*)d_in[3];
  const float* Ws    = (const float*)d_in[4];
  const float* bs    = (const float*)d_in[5];
  const float* Wout  = (const float*)d_in[6];
  const float* bout  = (const float*)d_in[7];
  float*       out   = (float*)d_out;

  char* ws = (char*)d_ws;
  size_t off = 0;
  auto alloc = [&](size_t bytes) -> char* {
    char* p = ws + off;
    off += (bytes + 255) & ~(size_t)255;
    return p;
  };
  float*   hbuf   = (float*)alloc((size_t)NNODES * HDIM * 4);
  float*   tbuf   = (float*)alloc((size_t)NNODES * HDIM * 4);
  float*   dinv   = (float*)alloc((size_t)NNODES * 4);
  float*   enorm  = (float*)alloc((size_t)NEDGES * 4);
  __bf16*  Wfrag  = (__bf16*)alloc((size_t)NLAYERS * HDIM * HDIM * 2);
  float*   pooled = (float*)alloc((size_t)NGRAPH * HDIM * 4);
  float*   counts = (float*)alloc((size_t)NGRAPH * 4);

  const int T = 256;
  const int NH = NNODES * HDIM;
  auto nb = [](int n, int t) { return (n + t - 1) / t; };

  // degree / norms
  k_fill<<<nb(NNODES, T), T, 0, stream>>>(dinv, 1.0f, NNODES);          // deg = 1
  k_count_deg<<<nb(NEDGES, T), T, 0, stream>>>(dst, dinv, NEDGES);
  k_dinv<<<nb(NNODES, T), T, 0, stream>>>(dinv, NNODES);                // deg -> rsqrt
  k_enorm<<<nb(NEDGES, T), T, 0, stream>>>(src, dst, dinv, enorm, NEDGES);

  // weights -> bf16 B-fragments; embedding lookup
  k_prep_w<<<nb(NLAYERS * 4 * 8 * 32 * 16, T), T, 0, stream>>>(Ws, Wfrag);
  k_init_h<<<nb(NH, T), T, 0, stream>>>(x, embed, hbuf, NH);

  // layers
  for (int l = 0; l < NLAYERS; ++l) {
    if (l == 0)
      k_gemm<false><<<nb(NNODES, 128), T, 0, stream>>>(hbuf, Wfrag, tbuf, NNODES);
    else
      k_gemm<true><<<nb(NNODES, 128), T, 0, stream>>>(hbuf, Wfrag + l * HDIM * HDIM,
                                                      tbuf, NNODES);
    k_init_agg<<<nb(NH, T), T, 0, stream>>>(tbuf, dinv, bs + l * HDIM, hbuf, NH);
    k_edge_scatter<<<nb(NEDGES * 32, T), T, 0, stream>>>(src, dst, enorm, tbuf,
                                                         hbuf, NEDGES);
  }

  // pooling + output head
  k_fill<<<nb(NGRAPH * HDIM, T), T, 0, stream>>>(pooled, 0.0f, NGRAPH * HDIM);
  k_fill<<<1, T, 0, stream>>>(counts, 0.0f, NGRAPH);
  k_pool<<<nb(NH, T), T, 0, stream>>>(hbuf, batch, pooled, counts, NH);
  k_out<<<1, T, 0, stream>>>(pooled, counts, Wout, bout, out);
}
